// DifferentialTransformerBlock_14121852469509
// MI455X (gfx1250) — compile-verified, matrix operand order checked by reading, not run on hardware
//
#include <hip/hip_runtime.h>
#include <hip/hip_bf16.h>

// ---------------------------------------------------------------------------
// Differential Transformer block for MI455X (gfx1250, wave32, WMMA).
// All heavy math through v_wmma_f32_16x16x32_f16 (f16 in, f32 accum).
// Shapes: B=2 N=2048 E=768 H=12 D=768 -> BN=4096, 2D=1536, HD=9216.
// V is produced d-major ([H][D][BN]) so attention stages it into LDS with
// contiguous 16B copies via the CDNA5 async global->LDS path (ASYNCcnt).
// ---------------------------------------------------------------------------

typedef __attribute__((ext_vector_type(16))) _Float16 v16h;
typedef __attribute__((ext_vector_type(8)))  float    v8f;
typedef __attribute__((ext_vector_type(4)))  int      v4i;

union Frag16 { v16h h; float4 f4[2]; };

#define WMMA_F16(Afrag, Bfrag, Cacc) \
    __builtin_amdgcn_wmma_f32_16x16x32_f16(false, (Afrag), false, (Bfrag), (short)0, (Cacc), false, false)

#if defined(__gfx1250__) && __has_builtin(__builtin_amdgcn_global_load_async_to_lds_b128) && \
    __has_builtin(__builtin_amdgcn_s_wait_asynccnt)
#define USE_ASYNC 1
#else
#define USE_ASYNC 0
#endif

#if USE_ASYNC
static __device__ __forceinline__ void async_copy16(const void* g, void* l) {
    __builtin_amdgcn_global_load_async_to_lds_b128(
        (__attribute__((address_space(1))) v4i*)(g),
        (__attribute__((address_space(3))) v4i*)(l), 0, 0);
}
static __device__ __forceinline__ void wait_async0() { __builtin_amdgcn_s_wait_asynccnt(0); }
#else
static __device__ __forceinline__ void async_copy16(const void* g, void* l) {
    *(uint4*)l = *(const uint4*)g;
}
static __device__ __forceinline__ void wait_async0() {}
#endif

static __device__ __forceinline__ void store_val(float* p, float v)     { *p = v; }
static __device__ __forceinline__ void store_val(_Float16* p, float v)  { *p = (_Float16)v; }

// ---------------------------------------------------------------------------
// f32 -> f16 conversion (weights)
// ---------------------------------------------------------------------------
__global__ __launch_bounds__(256) void f32_to_f16_kernel(const float* __restrict__ in,
                                                         _Float16* __restrict__ out, int n) {
    int i = blockIdx.x * blockDim.x + threadIdx.x;
    int stride = gridDim.x * blockDim.x;
    for (; i < n; i += stride) out[i] = (_Float16)in[i];
}

// ---------------------------------------------------------------------------
// RMSNorm (per row of E=768) fused with f16 conversion
// ---------------------------------------------------------------------------
__global__ __launch_bounds__(256) void rmsnorm_f16_kernel(const float* __restrict__ x,
                                                          const float* __restrict__ scale,
                                                          _Float16* __restrict__ y) {
    __shared__ float red[8];
    const int row = blockIdx.x;
    const float* xr = x + (size_t)row * 768;
    float ss = 0.f;
    for (int i = threadIdx.x; i < 768; i += 256) { float v = xr[i]; ss += v * v; }
    for (int off = 16; off; off >>= 1) ss += __shfl_xor(ss, off, 32);
    if ((threadIdx.x & 31) == 0) red[threadIdx.x >> 5] = ss;
    __syncthreads();
    float tot = 0.f;
#pragma unroll
    for (int i = 0; i < 8; ++i) tot += red[i];
    const float rms = sqrtf(tot / 768.0f);
    const float inv = 1.0f / (rms + 1e-8f);
    _Float16* yr = y + (size_t)row * 768;
    for (int i = threadIdx.x; i < 768; i += 256) yr[i] = (_Float16)(xr[i] * inv * scale[i]);
}

// ---------------------------------------------------------------------------
// LayerNorm * 0.95 + residual epilogue (per row of E=768)
// ---------------------------------------------------------------------------
__global__ __launch_bounds__(256) void ln_resid_kernel(const float* __restrict__ ao,
                                                       const float* __restrict__ lnw,
                                                       const float* __restrict__ lnb,
                                                       const float* __restrict__ resid,
                                                       float* __restrict__ out) {
    __shared__ float red1[8], red2[8];
    const int row = blockIdx.x;
    const float* ar = ao + (size_t)row * 768;
    float s = 0.f, s2 = 0.f;
    for (int i = threadIdx.x; i < 768; i += 256) { float v = ar[i]; s += v; s2 += v * v; }
    for (int off = 16; off; off >>= 1) { s += __shfl_xor(s, off, 32); s2 += __shfl_xor(s2, off, 32); }
    if ((threadIdx.x & 31) == 0) { red1[threadIdx.x >> 5] = s; red2[threadIdx.x >> 5] = s2; }
    __syncthreads();
    float ts = 0.f, ts2 = 0.f;
#pragma unroll
    for (int i = 0; i < 8; ++i) { ts += red1[i]; ts2 += red2[i]; }
    const float mu  = ts / 768.0f;
    const float var = ts2 / 768.0f - mu * mu;
    const float rstd = rsqrtf(var + 1e-5f);
    const float* rr = resid + (size_t)row * 768;
    float* orow = out + (size_t)row * 768;
    for (int i = threadIdx.x; i < 768; i += 256)
        orow[i] = ((ar[i] - mu) * rstd * lnw[i] + lnb[i]) * 0.95f + rr[i];
}

// ---------------------------------------------------------------------------
// Tiled f16 WMMA GEMM:  C = A[M,K] * B[K,N] + bias[N]
// Block tile 128x64, 8 waves, K stepped by 32, double-buffered LDS staging.
// A tile staged via async global->LDS when available; B staged transposed.
// TC=false: C[row*ldC+col] ; TC=true: C[col*ldC+row]  (d-major output for V).
// Batched over blockIdx.z.
// ---------------------------------------------------------------------------
template <typename OutT, bool TC>
__global__ __launch_bounds__(256) void gemm_f16_kernel(
    const _Float16* __restrict__ A, const _Float16* __restrict__ B,
    const float* __restrict__ bias, OutT* __restrict__ C,
    int M, int N, int K, long long ldC,
    long long sAb, long long sBb, long long sBiasb, long long sCb) {
    const int batch = blockIdx.z;
    A += (size_t)batch * sAb;
    B += (size_t)batch * sBb;
    C += (size_t)batch * sCb;
    const float* bi = bias + (size_t)batch * sBiasb;

    const int n0 = blockIdx.x * 64;
    const int m0 = blockIdx.y * 128;
    const int tid = threadIdx.x, lane = tid & 31, w = tid >> 5;
    const int kbase = (lane >> 4) << 3;          // 0 or 8 (f16 elems)
    const int lmod = lane & 15;

    __shared__ __align__(16) _Float16 lds_a[2][128 * 40];  // 128 rows x 32 f16 (pad->40)
    __shared__ __align__(16) _Float16 lds_b[2][64 * 40];   // 64 cols  x 32 f16 (transposed)

    auto issueA = [&](int k0, int buf) {
        for (int c = tid; c < 512; c += 256) {   // 512 x 16B
            int r = c >> 2, cc = c & 3;
            async_copy16(A + (size_t)(m0 + r) * K + k0 + cc * 8,
                         &lds_a[buf][r * 40 + cc * 8]);
        }
    };
    auto issueB = [&](int k0, int buf) {
        for (int u = tid; u < 1024; u += 256) {  // read [k][n] coalesced, store [n][k]
            int kk = u >> 5, nn = (u & 31) * 2;
            union { unsigned int u32; _Float16 hh[2]; } cv;
            cv.u32 = *(const unsigned int*)(B + (size_t)(k0 + kk) * N + n0 + nn);
            lds_b[buf][nn * 40 + kk]       = cv.hh[0];
            lds_b[buf][(nn + 1) * 40 + kk] = cv.hh[1];
        }
    };

    const int nk = K >> 5;
    issueA(0, 0);
    issueB(0, 0);
    wait_async0();
    __syncthreads();

    v8f acc[4] = {};
    for (int s = 0; s < nk; ++s) {
        const int buf = s & 1;
        if (s + 1 < nk) { issueA((s + 1) * 32, buf ^ 1); issueB((s + 1) * 32, buf ^ 1); }

        Frag16 af;
        const _Float16* ap = &lds_a[buf][(w * 16 + lmod) * 40 + kbase];
        af.f4[0] = *(const float4*)ap;
        af.f4[1] = *(const float4*)(ap + 16);
#pragma unroll
        for (int ct = 0; ct < 4; ++ct) {
            Frag16 bf;
            const _Float16* bp = &lds_b[buf][(ct * 16 + lmod) * 40 + kbase];
            bf.f4[0] = *(const float4*)bp;
            bf.f4[1] = *(const float4*)(bp + 16);
            acc[ct] = WMMA_F16(af.h, bf.h, acc[ct]);
        }
        wait_async0();     // next-stage async A complete
        __syncthreads();   // next-stage B stores visible; this-stage reads done
    }

    const int hi = (lane >> 4) << 3;
#pragma unroll
    for (int ct = 0; ct < 4; ++ct) {
        int col = n0 + ct * 16 + lmod;
        float bv = bi[col];
#pragma unroll
        for (int v = 0; v < 8; ++v) {
            int row = m0 + w * 16 + v + hi;
            float val = acc[ct][v] + bv;
            if (TC) store_val(C + (size_t)col * ldC + row, val);
            else    store_val(C + (size_t)row * ldC + col, val);
        }
    }
}

// ---------------------------------------------------------------------------
// Fused differential flash-attention.
// Grid: (N/16=128, H=12, B=2). Block: 256 threads = 8 waves.
// 16 query rows per block; waves split D=768 into 8x96-col slices.
// Two online softmaxes (Q1K1, Q2K2) accumulated into O1, O2 in f32,
// combined as O1/l1 - (e^lam + 0.05) * O2/l2 at the end.
// V tile (d-major global layout) staged to LDS via async global->LDS,
// overlapped with score reduction + softmax phases.
// ---------------------------------------------------------------------------
__global__ __launch_bounds__(256) void diff_attn_kernel(
    const _Float16* __restrict__ Qh,   // [H][BN][1536]
    const _Float16* __restrict__ Kh,   // [H][BN][1536]
    const _Float16* __restrict__ Vg,   // [H][768][BN]  (d-major)
    const float*    __restrict__ lam,  // [H]
    _Float16* __restrict__ Oc) {       // [BN][9216]  (b,n, h*768+d)
    __shared__ __align__(16) _Float16 Vt[768 * 32];     // V tile [d][m], 48KB
    __shared__ __align__(16) float    S[2][16][32];     // cross-wave score reduction
    __shared__ __align__(16) _Float16 P[2][16][32];     // softmax probabilities (f16)
    __shared__ float Ald[2][16];                        // per-row rescale factors
    __shared__ float Lld[2][16];                        // per-row running sums

    const int qb = blockIdx.x, h = blockIdx.y, b = blockIdx.z;
    const int tid = threadIdx.x, lane = tid & 31, w = tid >> 5;
    const int q0 = qb * 16;
    const int kbase = (lane >> 4) << 3;
    const int lmod = lane & 15;
    const size_t rowbase = (size_t)h * 4096 + (size_t)b * 2048;
    const size_t vbase   = (size_t)h * 768 * 4096 + (size_t)b * 2048;
    const float scale = 0.03608439182435161f;  // 1/sqrt(768)

    // Q fragments for this wave's d-slice, both branches: [br][kstep]
    Frag16 qf[2][3];
    {
        const size_t qrow = rowbase + q0 + lmod;
#pragma unroll
        for (int br = 0; br < 2; ++br)
#pragma unroll
            for (int kk = 0; kk < 3; ++kk) {
                const _Float16* qp = Qh + qrow * 1536 + br * 768 + w * 96 + kk * 32 + kbase;
                qf[br][kk].f4[0] = *(const float4*)qp;
                qf[br][kk].f4[1] = *(const float4*)(qp + 16);
            }
    }

    v8f acc1[6] = {}, acc2[6] = {};
    float m_r = -1e30f, l_r = 0.f;   // live only in wave 0

    for (int t = 0; t < 64; ++t) {
        const int k0 = t * 32;

        // ---- partial scores over this wave's d-slice (registers only) ----
        v8f s[2][2] = {};
        if (t < 63)  // prefetch next key tile row for this lane
            __builtin_prefetch(Kh + (rowbase + k0 + 32 + lmod) * 1536 + w * 96, 0, 1);
#pragma unroll
        for (int kk = 0; kk < 3; ++kk)
#pragma unroll
            for (int st = 0; st < 2; ++st) {
                const size_t krow = rowbase + k0 + st * 16 + lmod;
#pragma unroll
                for (int br = 0; br < 2; ++br) {
                    const _Float16* kp = Kh + krow * 1536 + br * 768 + w * 96 + kk * 32 + kbase;
                    Frag16 kf;
                    kf.f4[0] = *(const float4*)kp;
                    kf.f4[1] = *(const float4*)(kp + 16);
                    s[br][st] = WMMA_F16(qf[br][kk].h, kf.h, s[br][st]);
                }
            }

        __syncthreads();  // prior-iteration LDS consumers (P, Vt) done

        // ---- zero score buffer + issue async V-tile stage (contiguous 16B) ----
        {
            float* Sp = &S[0][0][0];
            for (int i = tid; i < 1024; i += 256) Sp[i] = 0.f;
        }
        for (int c = tid; c < 3072; c += 256) {  // 768 d-rows x 4 x 16B
            int d = c >> 2, q = (c & 3) * 8;
            async_copy16(Vg + vbase + (size_t)d * 4096 + k0 + q, &Vt[d * 32 + q]);
        }
        __syncthreads();  // S zeroed before atomics

        // ---- cross-wave reduction of partial scores ----
        {
            const int hi = (lane >> 4) << 3;
#pragma unroll
            for (int br = 0; br < 2; ++br)
#pragma unroll
                for (int st = 0; st < 2; ++st)
#pragma unroll
                    for (int v = 0; v < 8; ++v)
                        atomicAdd(&S[br][v + hi][st * 16 + lmod], s[br][st][v]);
        }
        __syncthreads();

        // ---- wave 0: two online softmaxes (lanes 0-15: br 0, 16-31: br 1) ----
        if (w == 0) {
            const int br = lane >> 4;
            const int r  = lane & 15;
            float mx = -1e30f;
            for (int k = 0; k < 32; ++k) mx = fmaxf(mx, S[br][r][k] * scale);
            const float m_new = fmaxf(m_r, mx);
            const float al = __expf(m_r - m_new);
            float psum = 0.f;
            for (int k = 0; k < 32; ++k) {
                float p = __expf(S[br][r][k] * scale - m_new);
                psum += p;
                P[br][r][k] = (_Float16)p;
            }
            l_r = l_r * al + psum;
            m_r = m_new;
            Ald[br][r] = al;
            Lld[br][r] = l_r;
        }
        wait_async0();    // V tile resident in LDS (overlapped with phases above)
        __syncthreads();

        // ---- rescale accumulators, then P*V WMMAs for this wave's d-slice ----
        {
            const int hi = (lane >> 4) << 3;
#pragma unroll
            for (int v = 0; v < 8; ++v) {
                float a1 = Ald[0][v + hi], a2 = Ald[1][v + hi];
#pragma unroll
                for (int tt = 0; tt < 6; ++tt) { acc1[tt][v] *= a1; acc2[tt][v] *= a2; }
            }
            Frag16 p1, p2;
            const _Float16* pp = &P[0][lmod][kbase];
            p1.f4[0] = *(const float4*)pp;
            p1.f4[1] = *(const float4*)(pp + 16);
            pp = &P[1][lmod][kbase];
            p2.f4[0] = *(const float4*)pp;
            p2.f4[1] = *(const float4*)(pp + 16);
#pragma unroll
            for (int tt = 0; tt < 6; ++tt) {
                const int d = w * 96 + tt * 16 + lmod;
                Frag16 vf;
                const _Float16* vp = &Vt[d * 32 + kbase];
                vf.f4[0] = *(const float4*)vp;
                vf.f4[1] = *(const float4*)(vp + 16);
                acc1[tt] = WMMA_F16(p1.h, vf.h, acc1[tt]);
                acc2[tt] = WMMA_F16(p2.h, vf.h, acc2[tt]);
            }
        }
    }
    __syncthreads();

    // ---- finalize: O = O1/l1 - lam_full * O2/l2, store f16 ----
    const float lamf = __expf(lam[h]) + 0.05f;
    const int hi = (lane >> 4) << 3;
#pragma unroll
    for (int v = 0; v < 8; ++v) {
        const int row = v + hi;
        const float il1 = 1.0f / Lld[0][row];
        const float il2 = 1.0f / Lld[1][row];
        const size_t grow = (size_t)b * 2048 + q0 + row;
#pragma unroll
        for (int tt = 0; tt < 6; ++tt) {
            const int col = h * 768 + w * 96 + tt * 16 + lmod;
            const float o = acc1[tt][v] * il1 - lamf * (acc2[tt][v] * il2);
            Oc[grow * 9216 + col] = (_Float16)o;
        }
    }
}

// ---------------------------------------------------------------------------
// Host orchestration
// ---------------------------------------------------------------------------
extern "C" void kernel_launch(void* const* d_in, const int* in_sizes, int n_in,
                              void* d_out, int out_size, void* d_ws, size_t ws_size,
                              hipStream_t stream) {
    (void)in_sizes; (void)n_in; (void)out_size; (void)ws_size;
    const float* x       = (const float*)d_in[0];
    const float* Wq      = (const float*)d_in[1];
    const float* bq      = (const float*)d_in[2];
    const float* Wk      = (const float*)d_in[3];
    const float* bk      = (const float*)d_in[4];
    const float* Wv      = (const float*)d_in[5];
    const float* bv      = (const float*)d_in[6];
    const float* lam     = (const float*)d_in[7];
    const float* Wo      = (const float*)d_in[8];
    const float* bo      = (const float*)d_in[9];
    const float* ln_w    = (const float*)d_in[10];
    const float* ln_b    = (const float*)d_in[11];
    const float* rms_sc  = (const float*)d_in[12];
    float* out = (float*)d_out;

    // workspace carve (256B-aligned slices)
    size_t off = 0;
    auto carve = [&](size_t bytes) -> void* {
        void* p = (char*)d_ws + off;
        off += (bytes + 255) & ~(size_t)255;
        return p;
    };
    const size_t nWq = 12ull * 768 * 1536, nWv = 12ull * 768 * 768, nWo = 9216ull * 768;
    _Float16* Wqh = (_Float16*)carve(nWq * 2);
    _Float16* Wkh = (_Float16*)carve(nWq * 2);
    _Float16* Wvh = (_Float16*)carve(nWv * 2);
    _Float16* Woh = (_Float16*)carve(nWo * 2);
    _Float16* Xn  = (_Float16*)carve(4096ull * 768 * 2);
    _Float16* Qhb = (_Float16*)carve(12ull * 4096 * 1536 * 2);
    _Float16* Khb = (_Float16*)carve(12ull * 4096 * 1536 * 2);
    _Float16* Vgb = (_Float16*)carve(12ull * 768 * 4096 * 2);   // d-major
    _Float16* Ocb = (_Float16*)carve(4096ull * 9216 * 2);
    float*    AO  = (float*)carve(4096ull * 768 * 4);
    float*    Ab  = (float*)carve(4096ull * 768 * 4);

    // weights -> f16
    f32_to_f16_kernel<<<2048, 256, 0, stream>>>(Wq, Wqh, (int)nWq);
    f32_to_f16_kernel<<<2048, 256, 0, stream>>>(Wk, Wkh, (int)nWq);
    f32_to_f16_kernel<<<2048, 256, 0, stream>>>(Wv, Wvh, (int)nWv);
    f32_to_f16_kernel<<<2048, 256, 0, stream>>>(Wo, Woh, (int)nWo);

    auto layer = [&](const float* X, float* Y) {
        rmsnorm_f16_kernel<<<4096, 256, 0, stream>>>(X, rms_sc, Xn);
        // Q, K: [4096,768] x [768,1536] per head
        gemm_f16_kernel<_Float16, false><<<dim3(24, 32, 12), 256, 0, stream>>>(
            Xn, Wqh, bq, Qhb, 4096, 1536, 768, 1536,
            0, 768ll * 1536, 1536, 4096ll * 1536);
        gemm_f16_kernel<_Float16, false><<<dim3(24, 32, 12), 256, 0, stream>>>(
            Xn, Wkh, bk, Khb, 4096, 1536, 768, 1536,
            0, 768ll * 1536, 1536, 4096ll * 1536);
        // V: [4096,768] x [768,768] per head, stored d-major [H][768][4096]
        gemm_f16_kernel<_Float16, true><<<dim3(12, 32, 12), 256, 0, stream>>>(
            Xn, Wvh, bv, Vgb, 4096, 768, 768, 4096,
            0, 768ll * 768, 768, 768ll * 4096);
        // fused differential attention
        diff_attn_kernel<<<dim3(128, 12, 2), 256, 0, stream>>>(Qhb, Khb, Vgb, lam, Ocb);
        // O-projection: [4096,9216] x [9216,768]
        gemm_f16_kernel<float, false><<<dim3(12, 32, 1), 256, 0, stream>>>(
            Ocb, Woh, bo, AO, 4096, 768, 9216, 768, 0, 0, 0, 0);
        // layernorm * (1 - lambda_init) + residual
        ln_resid_kernel<<<4096, 256, 0, stream>>>(AO, ln_w, ln_b, X, Y);
    };

    layer(x, Ab);        // a = mhda(rms(x)) + x
    layer(Ab, out);      // out = mhda(rms(a)) + a
}
